// IntTransformerLayer_27006754357350
// MI455X (gfx1250) — matrix operand order repaired and checked by reading, not compiled, not run on hardware
//
#include <hip/hip_runtime.h>
#include <hip/hip_bf16.h>
#include <math.h>

#define B_  1024
#define L_  200
#define D_  128
#define H_  4
#define FF_ 512
#define NL_ 2
#define DK_ 32

typedef _Float16 f16;
typedef _Float16 v8h  __attribute__((ext_vector_type(8)));
typedef _Float16 v16h __attribute__((ext_vector_type(16)));
typedef float    v8f  __attribute__((ext_vector_type(8)));

static __device__ inline v16h cat16(v8h lo, v8h hi) {
  v16h r;
#pragma unroll
  for (int i = 0; i < 8; ++i) { r[i] = lo[i]; r[i + 8] = hi[i]; }
  return r;
}
static __device__ inline v8h zero8h() {
  v8h z;
#pragma unroll
  for (int i = 0; i < 8; ++i) z[i] = (f16)0.0f;
  return z;
}
static __device__ inline v8f zero8f() {
  v8f z;
#pragma unroll
  for (int i = 0; i < 8; ++i) z[i] = 0.0f;
  return z;
}

// Wave32 butterfly reductions via ds_swizzle (group-of-32 xor mode:
// imm = {0[15], xor_mask[14:10], or_mask[9:5]=0, and_mask[4:0]=0x1f}).
#define SWZF(v, imm) \
  __int_as_float(__builtin_amdgcn_ds_swizzle(__float_as_int(v), (imm)))
static __device__ inline float wave_max(float v) {
  v = fmaxf(v, SWZF(v, 0x041f));
  v = fmaxf(v, SWZF(v, 0x081f));
  v = fmaxf(v, SWZF(v, 0x101f));
  v = fmaxf(v, SWZF(v, 0x201f));
  v = fmaxf(v, SWZF(v, 0x401f));
  return v;
}
static __device__ inline float wave_sum(float v) {
  v += SWZF(v, 0x041f);
  v += SWZF(v, 0x081f);
  v += SWZF(v, 0x101f);
  v += SWZF(v, 0x201f);
  v += SWZF(v, 0x401f);
  return v;
}

// A-layout fragment (16x32 f16): lane = row (lane&15), K pattern
// lanes 0-15: K {0..7,16..23}; lanes 16-31: K {8..15,24..31}.
static __device__ inline v16h load_a_frag(const f16* base, int stride, int lane) {
  const int m  = lane & 15;
  const int kb = (lane < 16) ? 0 : 8;
  const f16* p = base + m * stride + kb;
  v8h lo = *(const v8h*)(p);
  v8h hi = *(const v8h*)(p + 16);
  return cat16(lo, hi);
}
// B-layout fragment (32x16 f16): lane = output col (lane&15),
// lanes 0-15 hold K {0..7,16..23}, lanes 16-31 hold K {8..15,24..31} (matches A).
static __device__ inline v16h load_b_frag(const f16* base, int stride, int lane) {
  const int n  = lane & 15;
  const int kb = (lane < 16) ? 0 : 8;
  const f16* p = base + n * stride + kb;
  v8h lo = *(const v8h*)(p);
  v8h hi = *(const v8h*)(p + 16);
  return cat16(lo, hi);
}
static __device__ inline v8f wmma_f16(v16h a, v16h b, v8f c) {
  return __builtin_amdgcn_wmma_f32_16x16x32_f16(false, a, false, b, (short)0, c,
                                                false, false);
}

// ---------------------------------------------------------------- GEMM ------
// C[M,N] = A[M,K](f16,row-major) * Wt[N,K](f16) + bias, M%128==0, N%64==0, K%64==0
constexpr int EPI_F16    = 0;  // out f16
constexpr int EPI_ADD32  = 1;  // o32 += val (residual)
constexpr int EPI_GELU16 = 2;  // out f16 = gelu(val)
constexpr int EPI_F32    = 3;  // out f32

#define BM 128
#define BN 64
#define BK 64
#define ASTR 72
#define BSTR 72

template <int EPI>
__global__ __launch_bounds__(256) void gemm_f16(
    const f16* __restrict__ A, const f16* __restrict__ Wt,
    const float* __restrict__ bias, int M, int N, int K,
    float* __restrict__ o32, f16* __restrict__ o16) {
  __shared__ f16 As[BM * ASTR];
  __shared__ f16 Bs[BN * BSTR];
  const int t    = threadIdx.x;
  const int lane = t & 31;
  const int w    = t >> 5;
  const int m0   = blockIdx.y * BM;
  const int n0   = blockIdx.x * BN;
  const int wm   = (w & 3) * 32;
  const int wn   = (w >> 2) * 32;

  v8f acc[2][2];
#pragma unroll
  for (int i = 0; i < 2; ++i)
#pragma unroll
    for (int j = 0; j < 2; ++j) acc[i][j] = zero8f();

  for (int k0 = 0; k0 < K; k0 += BK) {
    // A tile 128x64 halves -> 1024 chunks of 8
#pragma unroll
    for (int it = 0; it < 4; ++it) {
      int idx = t + it * 256;
      int r = idx >> 3, c = (idx & 7) * 8;
      *(v8h*)&As[r * ASTR + c] = *(const v8h*)&A[(size_t)(m0 + r) * K + k0 + c];
    }
    // B tile 64x64 halves -> 512 chunks of 8
#pragma unroll
    for (int it = 0; it < 2; ++it) {
      int idx = t + it * 256;
      int r = idx >> 3, c = (idx & 7) * 8;
      *(v8h*)&Bs[r * BSTR + c] = *(const v8h*)&Wt[(size_t)(n0 + r) * K + k0 + c];
    }
    // Prefetch next K-panel toward cache while we compute this one.
    if (k0 + BK < K) {
      __builtin_prefetch(&A[(size_t)(m0 + (t & 127)) * K + k0 + BK], 0, 0);
      __builtin_prefetch(&Wt[(size_t)(n0 + (t & 63)) * K + k0 + BK], 0, 0);
    }
    __syncthreads();
#pragma unroll
    for (int ks = 0; ks < BK; ks += 32) {
      v16h af[2], bf[2];
#pragma unroll
      for (int i = 0; i < 2; ++i)
        af[i] = load_a_frag(&As[(wm + i * 16) * ASTR + ks], ASTR, lane);
#pragma unroll
      for (int j = 0; j < 2; ++j)
        bf[j] = load_b_frag(&Bs[(wn + j * 16) * BSTR + ks], BSTR, lane);
#pragma unroll
      for (int i = 0; i < 2; ++i)
#pragma unroll
        for (int j = 0; j < 2; ++j) acc[i][j] = wmma_f16(af[i], bf[j], acc[i][j]);
    }
    __syncthreads();
  }

  const int lrow = (lane < 16) ? 0 : 8;
  const int lcol = lane & 15;
#pragma unroll
  for (int i = 0; i < 2; ++i) {
#pragma unroll
    for (int j = 0; j < 2; ++j) {
      const int gn = n0 + wn + j * 16 + lcol;
      const float bv = bias ? bias[gn] : 0.0f;
#pragma unroll
      for (int r = 0; r < 8; ++r) {
        const int gm = m0 + wm + i * 16 + lrow + r;
        float v = acc[i][j][r] + bv;
        const size_t oi = (size_t)gm * N + gn;
        if (EPI == EPI_F16) {
          o16[oi] = (f16)v;
        } else if (EPI == EPI_ADD32) {
          o32[oi] += v;
        } else if (EPI == EPI_GELU16) {
          o16[oi] = (f16)(0.5f * v * (1.0f + erff(v * 0.70710678f)));
        } else {
          o32[oi] = v;
        }
      }
    }
  }
}

// ------------------------------------------------------------- attention ----
// Per (b, h, 32-query chunk): S = Q K^T (WMMA), mask+scale, softmax,
// O = P V (WMMA). Keys padded 200 -> 208 (S) -> 224 (P, K-mult-32).
#define SW   224
#define KST  40
#define VST  232
#define QROWS 32

__global__ __launch_bounds__(256) void attn_kernel(
    const f16* __restrict__ qb, const f16* __restrict__ kb,
    const f16* __restrict__ vb, const int* __restrict__ seqs,
    f16* __restrict__ ob) {
  __shared__ f16 Kl[208 * KST];
  __shared__ f16 Vt[32 * VST];
  __shared__ f16 Ql[QROWS * KST];
  __shared__ float Msk[SW];
  __shared__ float S[QROWS * SW];

  const int t    = threadIdx.x;
  const int lane = t & 31;
  const int w    = t >> 5;
  const int qc   = blockIdx.x;  // 0..6
  const int h    = blockIdx.y;
  const int b    = blockIdx.z;
  const size_t rowbase = (size_t)b * L_ * D_ + h * DK_;
  const int q0 = qc * QROWS;

  // Key-padding mask, hoisted once per block (removes global loads from the
  // score loop).
  for (int c = t; c < SW; c += 256) {
    const bool ok = (c < L_) && (seqs[(size_t)b * L_ + c] > 0);
    Msk[c] = ok ? 1.0f : -1.0f;
  }
  // K tile: 208 rows x 32 halves (zero rows >= L)
  for (int idx = t; idx < 208 * 4; idx += 256) {
    int r = idx >> 2, c = (idx & 3) * 8;
    v8h vv = (r < L_) ? *(const v8h*)&kb[rowbase + (size_t)r * D_ + c] : zero8h();
    *(v8h*)&Kl[r * KST + c] = vv;
  }
  // V^T tile [dk][key]: vectorized row reads, transposed 2-byte LDS stores.
  for (int idx = t; idx < SW * 4; idx += 256) {
    int key = idx >> 2, c = (idx & 3) * 8;
    v8h vv = (key < L_) ? *(const v8h*)&vb[rowbase + (size_t)key * D_ + c]
                        : zero8h();
#pragma unroll
    for (int u = 0; u < 8; ++u) Vt[(c + u) * VST + key] = vv[u];
  }
  // Q chunk: 32 rows x 32 halves
  if (t < QROWS * 4) {
    int r = t >> 2, c = (t & 3) * 8;
    v8h vv = (q0 + r < L_) ? *(const v8h*)&qb[rowbase + (size_t)(q0 + r) * D_ + c]
                           : zero8h();
    *(v8h*)&Ql[r * KST + c] = vv;
  }
  __syncthreads();

  // Phase 1: S tiles (2 row strips x 13 col tiles), K=32 single WMMA each
  {
    const int rstrip = (w & 1) * 16;
    const v16h af = load_a_frag(&Ql[rstrip * KST], KST, lane);
    const int lrow = (lane < 16) ? 0 : 8;
    for (int j = (w >> 1); j < 13; j += 4) {
      v16h bf = load_b_frag(&Kl[(j * 16) * KST], KST, lane);
      v8f c = zero8f();
      c = wmma_f16(af, bf, c);
      const int key = j * 16 + (lane & 15);
      const bool ok = Msk[key] > 0.0f;
#pragma unroll
      for (int r = 0; r < 8; ++r) {
        float v = ok ? c[r] * 0.17677669529f : -1e9f;
        S[(rstrip + lrow + r) * SW + key] = v;
      }
    }
  }
  __syncthreads();

  // Phase 2: softmax, 4 rows per wave; write P (f16) in place over S
  for (int rr = w * 4; rr < w * 4 + 4; ++rr) {
    float* Sr = &S[rr * SW];
    float mx = -1e30f;
    for (int c = lane; c < 208; c += 32) mx = fmaxf(mx, Sr[c]);
    mx = wave_max(mx);
    float sum = 0.0f;
    for (int c = lane; c < 208; c += 32) {
      float e = __expf(Sr[c] - mx);
      sum += e;
      Sr[c] = e;
    }
    sum = wave_sum(sum);
    const float inv = 1.0f / sum;
    f16* Pr = (f16*)Sr;
    for (int c = lane; c < SW; c += 32) {
      f16 pv = (f16)0.0f;
      if (c < 208) pv = (f16)(Sr[c] * inv);
      Pr[c] = pv;  // f16 write trails the f32 reads: no overlap hazard
    }
  }
  __syncthreads();

  // Phase 3: O = P(32x224) * V(224x32), 4 output tiles -> waves 0..3
  if (w < 4) {
    const int i = (w >> 1) * 16;
    const int j = (w & 1) * 16;
    v8f acc = zero8f();
    const int m = lane & 15;
    for (int kt = 0; kt < 7; ++kt) {
      const int kbp = ((lane < 16) ? 0 : 8) + kt * 32;
      const f16* Pr = (const f16*)&S[(i + m) * SW];
      v8h lo = *(const v8h*)(Pr + kbp);
      v8h hi = *(const v8h*)(Pr + kbp + 16);
      v16h af = cat16(lo, hi);
      v16h bf = load_b_frag(&Vt[j * VST + kt * 32], VST, lane);
      acc = wmma_f16(af, bf, acc);
    }
    const int lrow = (lane < 16) ? 0 : 8;
    const int lcol = lane & 15;
#pragma unroll
    for (int r = 0; r < 8; ++r) {
      const int ql = q0 + i + lrow + r;
      if (ql < L_)
        ob[(size_t)b * L_ * D_ + (size_t)ql * D_ + h * DK_ + j + lcol] = (f16)acc[r];
    }
  }
}

// -------------------------------------------------------------- small ops ---
__global__ __launch_bounds__(128) void ln_kernel(
    const float* __restrict__ x, const float* __restrict__ wln,
    const float* __restrict__ bln, f16* __restrict__ out) {
  __shared__ float red[4];
  const int t = threadIdx.x;
  const size_t row = blockIdx.x;
  const float v = x[row * D_ + t];
  float s = wave_sum(v);
  if ((t & 31) == 0) red[t >> 5] = s;
  __syncthreads();
  const float mu = (red[0] + red[1] + red[2] + red[3]) * (1.0f / D_);
  __syncthreads();
  const float d = v - mu;
  float s2 = wave_sum(d * d);
  if ((t & 31) == 0) red[t >> 5] = s2;
  __syncthreads();
  const float var = (red[0] + red[1] + red[2] + red[3]) * (1.0f / D_);
  const float r = rsqrtf(var + 1e-5f);
  out[row * D_ + t] = (f16)(d * r * wln[t] + bln[t]);
}

__global__ void embed_kernel(const int* __restrict__ seqs,
                             const float* __restrict__ tok,
                             const float* __restrict__ pos,
                             float* __restrict__ x) {
  const size_t row = blockIdx.x;
  const int t = threadIdx.x;
  const int id = seqs[row];
  const int l = (int)(row % L_);
  x[row * D_ + t] = tok[(size_t)id * D_ + t] + pos[(size_t)l * D_ + t];
}

__global__ void transpose_f32_f16(const float* __restrict__ in,
                                  f16* __restrict__ out, int K, int N) {
  const size_t i = (size_t)blockIdx.x * blockDim.x + threadIdx.x;
  if (i < (size_t)K * N) {
    const int k = (int)(i / N), n = (int)(i % N);
    out[(size_t)n * K + k] = (f16)in[i];
  }
}

__global__ void f32_to_f16(const float* __restrict__ in, f16* __restrict__ out,
                           size_t n) {
  const size_t i = (size_t)blockIdx.x * blockDim.x + threadIdx.x;
  if (i < n) out[i] = (f16)in[i];
}

// ----------------------------------------------------------------- driver ---
extern "C" void kernel_launch(void* const* d_in, const int* in_sizes, int n_in,
                              void* d_out, int out_size, void* d_ws,
                              size_t ws_size, hipStream_t stream) {
  (void)in_sizes; (void)n_in; (void)out_size; (void)ws_size;
  const int*   seqs = (const int*)d_in[0];
  const float* tok  = (const float*)d_in[1];
  const float* pos  = (const float*)d_in[2];
  const float* Wq   = (const float*)d_in[3];
  const float* bq   = (const float*)d_in[4];
  const float* Wk   = (const float*)d_in[5];
  const float* bk   = (const float*)d_in[6];
  const float* Wv   = (const float*)d_in[7];
  const float* bv   = (const float*)d_in[8];
  const float* Wo   = (const float*)d_in[9];
  const float* bo   = (const float*)d_in[10];
  const float* ln1w = (const float*)d_in[11];
  const float* ln1b = (const float*)d_in[12];
  const float* ln2w = (const float*)d_in[13];
  const float* ln2b = (const float*)d_in[14];
  const float* W1   = (const float*)d_in[15];
  const float* b1   = (const float*)d_in[16];
  const float* W2   = (const float*)d_in[17];
  const float* b2   = (const float*)d_in[18];
  const float* fcW  = (const float*)d_in[19];
  const float* fcb  = (const float*)d_in[20];
  float* out = (float*)d_out;

  char* ws = (char*)d_ws;
  size_t off = 0;
  auto alloc = [&](size_t bytes) -> void* {
    void* p = ws + off;
    off = (off + bytes + 255) & ~(size_t)255;
    return p;
  };
  const size_t BL = (size_t)B_ * L_;
  float* x   = (float*)alloc(BL * D_ * 4);
  f16* hb    = (f16*)alloc(BL * D_ * 2);
  f16* qbuf  = (f16*)alloc(BL * D_ * 2);
  f16* kbuf  = (f16*)alloc(BL * D_ * 2);
  f16* vbuf  = (f16*)alloc(BL * D_ * 2);
  f16* abuf  = (f16*)alloc(BL * D_ * 2);
  f16* ffb   = (f16*)alloc(BL * FF_ * 2);
  f16* Wqt   = (f16*)alloc((size_t)NL_ * D_ * D_ * 2);
  f16* Wkt   = (f16*)alloc((size_t)NL_ * D_ * D_ * 2);
  f16* Wvt   = (f16*)alloc((size_t)NL_ * D_ * D_ * 2);
  f16* Wot   = (f16*)alloc((size_t)NL_ * D_ * D_ * 2);
  f16* W1t   = (f16*)alloc((size_t)NL_ * FF_ * D_ * 2);
  f16* W2t   = (f16*)alloc((size_t)NL_ * D_ * FF_ * 2);
  f16* fcWt  = (f16*)alloc((size_t)D_ * L_ * D_ * 2);

  auto tg = [](size_t n) { return (unsigned)((n + 255) / 256); };

  // Weight prep: f32 [K][N] -> f16 [N][K]
  for (int i = 0; i < NL_; ++i) {
    const size_t dd = (size_t)i * D_ * D_;
    const size_t df = (size_t)i * D_ * FF_;
    transpose_f32_f16<<<tg((size_t)D_ * D_), 256, 0, stream>>>(Wq + dd, Wqt + dd, D_, D_);
    transpose_f32_f16<<<tg((size_t)D_ * D_), 256, 0, stream>>>(Wk + dd, Wkt + dd, D_, D_);
    transpose_f32_f16<<<tg((size_t)D_ * D_), 256, 0, stream>>>(Wv + dd, Wvt + dd, D_, D_);
    transpose_f32_f16<<<tg((size_t)D_ * D_), 256, 0, stream>>>(Wo + dd, Wot + dd, D_, D_);
    transpose_f32_f16<<<tg((size_t)D_ * FF_), 256, 0, stream>>>(W1 + df, W1t + df, D_, FF_);
    transpose_f32_f16<<<tg((size_t)FF_ * D_), 256, 0, stream>>>(W2 + df, W2t + df, FF_, D_);
  }
  transpose_f32_f16<<<tg((size_t)L_ * D_ * D_), 256, 0, stream>>>(fcW, fcWt, L_ * D_, D_);

  embed_kernel<<<(unsigned)BL, D_, 0, stream>>>(seqs, tok, pos, x);

  const dim3 gD(D_ / BN, (unsigned)(BL / BM));   // 2 x 1600
  const dim3 gF(FF_ / BN, (unsigned)(BL / BM));  // 8 x 1600
  for (int i = 0; i < NL_; ++i) {
    const size_t dd = (size_t)i * D_ * D_;
    const size_t df = (size_t)i * D_ * FF_;
    ln_kernel<<<(unsigned)BL, D_, 0, stream>>>(x, ln1w + i * D_, ln1b + i * D_, hb);
    gemm_f16<EPI_F16><<<gD, 256, 0, stream>>>(hb, Wqt + dd, bq + i * D_, (int)BL, D_, D_, nullptr, qbuf);
    gemm_f16<EPI_F16><<<gD, 256, 0, stream>>>(hb, Wkt + dd, bk + i * D_, (int)BL, D_, D_, nullptr, kbuf);
    gemm_f16<EPI_F16><<<gD, 256, 0, stream>>>(hb, Wvt + dd, bv + i * D_, (int)BL, D_, D_, nullptr, vbuf);
    attn_kernel<<<dim3(7, H_, B_), 256, 0, stream>>>(qbuf, kbuf, vbuf, seqs, abuf);
    gemm_f16<EPI_ADD32><<<gD, 256, 0, stream>>>(abuf, Wot + dd, bo + i * D_, (int)BL, D_, D_, x, nullptr);
    ln_kernel<<<(unsigned)BL, D_, 0, stream>>>(x, ln2w + i * D_, ln2b + i * D_, hb);
    gemm_f16<EPI_GELU16><<<gF, 256, 0, stream>>>(hb, W1t + df, b1 + i * FF_, (int)BL, FF_, D_, nullptr, ffb);
    gemm_f16<EPI_ADD32><<<gD, 256, 0, stream>>>(ffb, W2t + df, b2 + i * D_, (int)BL, D_, FF_, x, nullptr);
  }
  f32_to_f16<<<tg(BL * D_), 256, 0, stream>>>(x, hb, BL * D_);
  const dim3 gO(D_ / BN, B_ / BM);  // 2 x 8
  gemm_f16<EPI_F32><<<gO, 256, 0, stream>>>(hb, fcWt, fcb, B_, D_, L_ * D_, out, nullptr);
}
